// AttentionLayer_26508538151629
// MI455X (gfx1250) — matrix operand order repaired and checked by reading, not compiled, not run on hardware
//
#include <hip/hip_runtime.h>
#include <hip/hip_bf16.h>

// ---------------------------------------------------------------------------
// MI455X (gfx1250, wave32) implementation. All heavy math through
// v_wmma_f32_16x16x32_bf16. Attention is fused (flash-style) so the
// 8x2304x2304 energy matrix never touches HBM. Conv GEMMs are LDS
// double-buffered with hoisted im2col math + weight prefetch; softmax uses
// all 256 threads with in-wave shuffle reductions.
// Workspace use: ~75 MB (bf16 copies + fp32/bf16 activations).
// ---------------------------------------------------------------------------

typedef __attribute__((ext_vector_type(16))) __bf16 v16bf;
typedef __attribute__((ext_vector_type(8)))  __bf16 bf16x8;
typedef __attribute__((ext_vector_type(8)))  float  v8f;

#define NSP 2304   // 48*48
#define HW_ 48

// ---- WMMA fragment helpers (layouts per CDNA5 ISA 7.12.2) ------
// A (16x32 bf16): lane lo=row, per-lane element e -> K = (e/8)*16 + hi*8 + e%8
static __device__ __forceinline__ v16bf frag_a(const __bf16* rowp, int hi) {
  union { v16bf v; bf16x8 h[2]; } u;
  u.h[0] = *(const bf16x8*)(rowp + hi * 8);
  u.h[1] = *(const bf16x8*)(rowp + 16 + hi * 8);
  return u.v;
}
// B (32x16 bf16): lane lo=column, per-lane element e -> K = (e/8)*16 + hi*8 + e%8
static __device__ __forceinline__ v16bf frag_b(const __bf16* colp, int hi) {
  union { v16bf v; bf16x8 h[2]; } u;
  u.h[0] = *(const bf16x8*)(colp + hi * 8);
  u.h[1] = *(const bf16x8*)(colp + 16 + hi * 8);
  return u.v;
}
static __device__ __forceinline__ v8f wmma_bf16(v16bf a, v16bf b, v8f c) {
  return __builtin_amdgcn_wmma_f32_16x16x32_bf16(false, a, false, b,
                                                 (short)0, c, false, false);
}
static __device__ __forceinline__ v8f zero8() {
  v8f z;
#pragma unroll
  for (int i = 0; i < 8; ++i) z[i] = 0.0f;
  return z;
}

// ---------------------------------------------------------------------------
// fp32 -> bf16 elementwise convert
// ---------------------------------------------------------------------------
__global__ __launch_bounds__(256) void k_f32_to_bf16(const float* __restrict__ src,
                                                     __bf16* __restrict__ dst,
                                                     long long n) {
  long long i = (long long)blockIdx.x * 256 + threadIdx.x;
  long long stride = (long long)gridDim.x * 256;
  for (; i < n; i += stride) dst[i] = (__bf16)src[i];
}

// ---------------------------------------------------------------------------
// Fold conv-bias + BN into per-channel scale/bias: y = relu(conv*scale+bias)
// ---------------------------------------------------------------------------
__global__ __launch_bounds__(256) void k_bn_fold(const float* __restrict__ cb,
                                                 const float* __restrict__ g,
                                                 const float* __restrict__ be,
                                                 const float* __restrict__ m,
                                                 const float* __restrict__ v,
                                                 float* __restrict__ scale,
                                                 float* __restrict__ bias, int n) {
  int i = blockIdx.x * 256 + threadIdx.x;
  if (i < n) {
    float inv = g[i] * rsqrtf(v[i] + 1e-5f);
    scale[i] = inv;
    bias[i] = cb[i] * inv + be[i] - m[i] * inv;
  }
}

// ---------------------------------------------------------------------------
// 3x3 conv as implicit GEMM:  Y[co][j] = sum_k W[co][k] * im2col(X)[k][j]
// K = Cin*9. Tile 128(M) x 64(N), 8 waves, each wave 32x32 (2x2 wmma tiles).
// LDS double-buffered; im2col (h,w) bases hoisted out of the k-loop.
// Fused BN scale/bias + ReLU. Writes fp32 and/or bf16.
// ---------------------------------------------------------------------------
__global__ __launch_bounds__(256)
void k_conv3x3_wmma(const __bf16* __restrict__ Wb,   // [Cout][Cin*9]
                    const __bf16* __restrict__ Xb,   // [B][Cin][NSP]
                    const float* __restrict__ scale, // [Cout]
                    const float* __restrict__ bias,  // [Cout]
                    float* __restrict__ Yf,          // [B][Cout][NSP] or null
                    __bf16* __restrict__ Yb,         // [B][Cout][NSP] or null
                    int Cin, int Cout) {
  const int K  = Cin * 9;
  const int b  = blockIdx.z;
  const int m0 = blockIdx.x * 128;
  const int j0 = blockIdx.y * 64;
  const __bf16* X = Xb + (size_t)b * Cin * NSP;

  __shared__ __align__(16) __bf16 As[2][128][40];  // [buf][row][K]
  __shared__ __align__(16) __bf16 Bs[2][64][40];   // [buf][col][K] (transposed)

  const int tid = threadIdx.x;
  const int wave = tid >> 5, lane = tid & 31;
  const int lo = lane & 15, hi = lane >> 4;
  const int mw = wave & 3, nw = wave >> 2;         // 4x2 wave grid

  // --- per-thread loader coordinates (loop-invariant) ---
  const int a_row = tid >> 1;
  const int a_ks  = (tid & 1) * 16;
  const __bf16* a_src = Wb + (size_t)(m0 + a_row) * K + a_ks;

  const int b_k  = tid & 31;
  const int b_cb = (tid >> 5) * 8;
  int hbase[8], wbase[8];
#pragma unroll
  for (int i = 0; i < 8; ++i) {
    const int j  = j0 + b_cb + i;
    const int hq = j / HW_;
    hbase[i] = hq;
    wbase[i] = j - hq * HW_;
  }

  v8f acc[2][2];
#pragma unroll
  for (int mt = 0; mt < 2; ++mt)
#pragma unroll
    for (int nt = 0; nt < 2; ++nt) acc[mt][nt] = zero8();

  auto stage = [&](int k0, int buf) {
    // A: 128x32 weights, 32 bf16-bytes per thread, coalesced
    *(bf16x8*)&As[buf][a_row][a_ks]     = *(const bf16x8*)(a_src + k0);
    *(bf16x8*)&As[buf][a_row][a_ks + 8] = *(const bf16x8*)(a_src + k0 + 8);
    if (k0 + 64 < K)  // warm L2/GL1 for the tile after next
      __builtin_prefetch(a_src + k0 + 64, 0, 1);
    // B: im2col gather, stored transposed [col][K]
    const int kk = k0 + b_k;
    const int ci = kk / 9;
    const int r  = kk - ci * 9;
    const int r3 = r / 3;
    const int kh = r3 - 1;
    const int kw = r - r3 * 3 - 1;
    const __bf16* xc = X + (size_t)ci * NSP;
#pragma unroll
    for (int i = 0; i < 8; ++i) {
      const int hh = hbase[i] + kh;
      const int ww = wbase[i] + kw;
      __bf16 val = (__bf16)0.0f;
      if ((unsigned)hh < (unsigned)HW_ && (unsigned)ww < (unsigned)HW_)
        val = xc[hh * HW_ + ww];
      Bs[buf][b_cb + i][b_k] = val;
    }
  };

  stage(0, 0);
  int cur = 0;
  for (int k0 = 0; k0 < K; k0 += 32) {
    __syncthreads();                 // staged tile visible to all waves
    if (k0 + 32 < K) stage(k0 + 32, cur ^ 1);  // fill other buffer meanwhile

    v16bf bfr[2];
#pragma unroll
    for (int nt = 0; nt < 2; ++nt)
      bfr[nt] = frag_b(&Bs[cur][nw * 32 + nt * 16 + lo][0], hi);
#pragma unroll
    for (int mt = 0; mt < 2; ++mt) {
      v16bf afr = frag_a(&As[cur][mw * 32 + mt * 16 + lo][0], hi);
#pragma unroll
      for (int nt = 0; nt < 2; ++nt)
        acc[mt][nt] = wmma_bf16(afr, bfr[nt], acc[mt][nt]);
    }
    cur ^= 1;
  }

  // ---- epilogue: BN + ReLU --------------------------------------------
#pragma unroll
  for (int mt = 0; mt < 2; ++mt) {
#pragma unroll
    for (int nt = 0; nt < 2; ++nt) {
      const int jj = j0 + nw * 32 + nt * 16 + lo;
#pragma unroll
      for (int g = 0; g < 8; ++g) {
        const int co = m0 + mw * 32 + mt * 16 + g + 8 * hi;
        float val = acc[mt][nt][g] * scale[co] + bias[co];
        val = fmaxf(val, 0.0f);
        const size_t off = ((size_t)b * Cout + co) * NSP + jj;
        if (Yf) Yf[off] = val;
        if (Yb) Yb[off] = (__bf16)val;
      }
    }
  }
}

// ---------------------------------------------------------------------------
// 1x1 conv as GEMM: Y[m][j] = sum_k W[m][k] * X[k][j] + bias[m]  (bf16 out)
// Tile 64(M) x 64(N); 8 waves, each 16x32 (1x2 tiles).
// ---------------------------------------------------------------------------
__global__ __launch_bounds__(256)
void k_gemm1x1_wmma(const __bf16* __restrict__ Wb,  // [M][K]
                    const __bf16* __restrict__ Xb,  // [B][K][NSP]
                    const float* __restrict__ bias, // [M]
                    __bf16* __restrict__ Yb,        // [B][M][NSP]
                    int M, int Kd) {
  const int b  = blockIdx.z;
  const int m0 = blockIdx.x * 64;
  const int j0 = blockIdx.y * 64;
  const __bf16* X = Xb + (size_t)b * Kd * NSP;

  __shared__ __align__(16) __bf16 As[2][64][40];
  __shared__ __align__(16) __bf16 Bs[2][64][40];  // [col][K]

  const int tid = threadIdx.x;
  const int wave = tid >> 5, lane = tid & 31;
  const int lo = lane & 15, hi = lane >> 4;
  const int mw = wave & 3, nw = wave >> 2;

  const int a_row = tid >> 2;
  const int a_ks  = (tid & 3) * 8;
  const __bf16* a_src = Wb + (size_t)(m0 + a_row) * Kd + a_ks;
  const int b_k  = tid & 31;
  const int b_cb = (tid >> 5) * 8;
  const __bf16* b_src = X + (size_t)b_k * NSP + j0 + b_cb;

  v8f acc[2] = {zero8(), zero8()};

  auto stage = [&](int k0, int buf) {
    *(bf16x8*)&As[buf][a_row][a_ks] = *(const bf16x8*)(a_src + k0);
    bf16x8 v = *(const bf16x8*)(b_src + (size_t)k0 * NSP);
#pragma unroll
    for (int i = 0; i < 8; ++i) Bs[buf][b_cb + i][b_k] = v[i];
  };

  stage(0, 0);
  int cur = 0;
  for (int k0 = 0; k0 < Kd; k0 += 32) {
    __syncthreads();
    if (k0 + 32 < Kd) stage(k0 + 32, cur ^ 1);
    v16bf afr = frag_a(&As[cur][mw * 16 + lo][0], hi);
#pragma unroll
    for (int nt = 0; nt < 2; ++nt) {
      v16bf bfr = frag_b(&Bs[cur][nw * 32 + nt * 16 + lo][0], hi);
      acc[nt] = wmma_bf16(afr, bfr, acc[nt]);
    }
    cur ^= 1;
  }

#pragma unroll
  for (int nt = 0; nt < 2; ++nt) {
    const int jj = j0 + nw * 32 + nt * 16 + lo;
#pragma unroll
    for (int g = 0; g < 8; ++g) {
      const int mrow = m0 + mw * 16 + g + 8 * hi;
      float val = acc[nt][g] + bias[mrow];
      Yb[((size_t)b * M + mrow) * NSP + jj] = (__bf16)val;
    }
  }
}

// ---------------------------------------------------------------------------
// Fused flash attention + residual:
//   S = q^T k  (d=64), attn = softmax(S) (online), O = attn * v^T,
//   y2 = bf16(gamma * O + x1)
// One block = (batch, 32 query rows). 8 waves:
//   S phase : wave w -> tile (w&1, w>>1) of the 32x64 score block
//   softmax : 8 threads per row (all 256 threads), shuffle reductions
//   PV phase: wave w -> O rows (w&1)*16, channel cols (w>>1)*64 (4 tiles)
// ---------------------------------------------------------------------------
__global__ __launch_bounds__(256)
void k_attn_flash(const __bf16* __restrict__ Qb,  // [B][64][NSP]
                  const __bf16* __restrict__ Kb,  // [B][64][NSP]
                  const __bf16* __restrict__ Vb,  // [B][256][NSP]
                  const float* __restrict__ X1f,  // [B][256][NSP]
                  const float* __restrict__ gamma,
                  __bf16* __restrict__ Y2b) {     // [B][256][NSP]
  const int b  = blockIdx.y;
  const int i0 = blockIdx.x * 32;

  __shared__ __align__(16) __bf16 qT[32][64];   // [row i][K=c]
  __shared__ __align__(16) __bf16 kT[64][72];   // [col j][K=c]
  __shared__ __align__(16) __bf16 pT[32][72];   // [row i][K=j]
  __shared__ __align__(16) __bf16 vS[256][72];  // [col c][K=j]
  __shared__ float sS[32][64];
  __shared__ float fac[32];
  __shared__ float lrow[32];

  const int tid = threadIdx.x;
  const int wave = tid >> 5, lane = tid & 31;
  const int lo = lane & 15, hi = lane >> 4;

  const __bf16* Q = Qb + (size_t)b * 64 * NSP;
  const __bf16* K = Kb + (size_t)b * 64 * NSP;
  const __bf16* V = Vb + (size_t)b * 256 * NSP;

  // stage q^T once: qT[i][c] = Q[c][i0+i]
  {
    const int i  = tid & 31;
    const int cb = (tid >> 5) * 8;
#pragma unroll
    for (int c = 0; c < 8; ++c)
      qT[i][cb + c] = Q[(size_t)(cb + c) * NSP + i0 + i];
  }

  const int mtS = wave & 1, ntS = wave >> 1;   // score-tile assignment
  const int mtO = wave & 1, cb0 = (wave >> 1) * 64;
  const int srow = tid >> 3, ssub = tid & 7;   // softmax: 8 threads / row

  v8f oacc[4];
#pragma unroll
  for (int t = 0; t < 4; ++t) oacc[t] = zero8();
  float mrun = -INFINITY, lsum = 0.0f;  // per-row stats, replicated x8

  for (int j0 = 0; j0 < NSP; j0 += 64) {
    __syncthreads();  // previous PV done before restaging kT/vS
    // stage kT[j][c] (coalesced global, transposed LDS store)
    {
      const int c  = tid >> 2;
      const int jb = (tid & 3) * 16;
      const __bf16* src = K + (size_t)c * NSP + j0 + jb;
#pragma unroll
      for (int j = 0; j < 16; ++j) kT[jb + j][c] = src[j];
    }
    // stage vS[c][j] (natural layout: contiguous both sides)
    {
      const __bf16* src = V + (size_t)tid * NSP + j0;
#pragma unroll
      for (int j = 0; j < 64; j += 8)
        *(bf16x8*)&vS[tid][j] = *(const bf16x8*)(src + j);
    }
    __syncthreads();

    // ---- S = qT x kT (one 16x16 tile/wave, K=64 = 2 wmma) ----------------
    v8f s = zero8();
#pragma unroll
    for (int ks = 0; ks < 2; ++ks) {
      v16bf afr = frag_a(&qT[mtS * 16 + lo][ks * 32], hi);
      v16bf bfr = frag_b(&kT[ntS * 16 + lo][ks * 32], hi);
      s = wmma_bf16(afr, bfr, s);
    }
#pragma unroll
    for (int g = 0; g < 8; ++g)
      sS[mtS * 16 + g + 8 * hi][ntS * 16 + lo] = s[g];
    __syncthreads();

    // ---- online softmax: 8 threads per row, butterfly reductions ---------
    {
      float sv[8];
      const float4* rp = (const float4*)&sS[srow][ssub * 8];
      float4 x0 = rp[0], x1 = rp[1];
      sv[0]=x0.x; sv[1]=x0.y; sv[2]=x0.z; sv[3]=x0.w;
      sv[4]=x1.x; sv[5]=x1.y; sv[6]=x1.z; sv[7]=x1.w;
      float mx = sv[0];
#pragma unroll
      for (int e = 1; e < 8; ++e) mx = fmaxf(mx, sv[e]);
#pragma unroll
      for (int msk = 1; msk < 8; msk <<= 1)
        mx = fmaxf(mx, __shfl_xor(mx, msk, 32));
      mx = fmaxf(mx, mrun);
      const float f = __expf(mrun - mx);
      float sum = 0.0f;
      union { bf16x8 v; __bf16 e[8]; } pb;
#pragma unroll
      for (int e = 0; e < 8; ++e) {
        const float pe = __expf(sv[e] - mx);
        sum += pe;
        pb.e[e] = (__bf16)pe;
      }
      *(bf16x8*)&pT[srow][ssub * 8] = pb.v;   // aligned b128 store
#pragma unroll
      for (int msk = 1; msk < 8; msk <<= 1)
        sum += __shfl_xor(sum, msk, 32);
      lsum = lsum * f + sum;
      mrun = mx;
      if (ssub == 0) fac[srow] = f;
    }
    __syncthreads();

    // ---- rescale O, then O += P x V^T ------------------------------------
    float fg[8];
#pragma unroll
    for (int g = 0; g < 8; ++g) fg[g] = fac[mtO * 16 + g + 8 * hi];
#pragma unroll
    for (int ct = 0; ct < 4; ++ct)
#pragma unroll
      for (int g = 0; g < 8; ++g) oacc[ct][g] *= fg[g];
#pragma unroll
    for (int ks = 0; ks < 2; ++ks) {
      v16bf afr = frag_a(&pT[mtO * 16 + lo][ks * 32], hi);
#pragma unroll
      for (int ct = 0; ct < 4; ++ct) {
        v16bf bfr = frag_b(&vS[cb0 + ct * 16 + lo][ks * 32], hi);
        oacc[ct] = wmma_bf16(afr, bfr, oacc[ct]);
      }
    }
  }

  if (ssub == 0) lrow[srow] = lsum;
  __syncthreads();

  const float g0 = gamma[0];
  const float* x1 = X1f + (size_t)b * 256 * NSP;
  __bf16* y2 = Y2b + (size_t)b * 256 * NSP;
#pragma unroll
  for (int ct = 0; ct < 4; ++ct) {
    const int c = cb0 + ct * 16 + lo;
#pragma unroll
    for (int g = 0; g < 8; ++g) {
      const int i = i0 + mtO * 16 + g + 8 * hi;
      const float inv_l = 1.0f / lrow[mtO * 16 + g + 8 * hi];
      const float val = g0 * oacc[ct][g] * inv_l + x1[(size_t)c * NSP + i];
      y2[(size_t)c * NSP + i] = (__bf16)val;
    }
  }
}

// ---------------------------------------------------------------------------
// Launch
// ---------------------------------------------------------------------------
extern "C" void kernel_launch(void* const* d_in, const int* in_sizes, int n_in,
                              void* d_out, int out_size, void* d_ws, size_t ws_size,
                              hipStream_t stream) {
  const int B = 8, Cin = 512, Cout = 256, Cqk = 64;

  const float* x     = (const float*)d_in[0];
  const float* w_pre = (const float*)d_in[1];
  const float* b_pre = (const float*)d_in[2];
  const float* bn1_g = (const float*)d_in[3];
  const float* bn1_b = (const float*)d_in[4];
  const float* bn1_m = (const float*)d_in[5];
  const float* bn1_v = (const float*)d_in[6];
  const float* w_q   = (const float*)d_in[7];
  const float* b_q   = (const float*)d_in[8];
  const float* w_k   = (const float*)d_in[9];
  const float* b_k   = (const float*)d_in[10];
  const float* w_v   = (const float*)d_in[11];
  const float* b_v   = (const float*)d_in[12];
  const float* w_f   = (const float*)d_in[13];
  const float* b_f   = (const float*)d_in[14];
  const float* bn2_g = (const float*)d_in[15];
  const float* bn2_b = (const float*)d_in[16];
  const float* bn2_m = (const float*)d_in[17];
  const float* bn2_v = (const float*)d_in[18];
  const float* gamma = (const float*)d_in[19];

  // ---- workspace carve-out (deterministic, ~75 MB) ----
  char* p = (char*)d_ws;
  auto alloc = [&](size_t bytes) -> void* {
    void* r = (void*)p;
    p += (bytes + 255) & ~(size_t)255;
    return r;
  };
  __bf16* xb   = (__bf16*)alloc((size_t)B * Cin * NSP * 2);
  __bf16* wpre = (__bf16*)alloc((size_t)Cout * Cin * 9 * 2);
  __bf16* wqb  = (__bf16*)alloc((size_t)Cqk * Cout * 2);
  __bf16* wkb  = (__bf16*)alloc((size_t)Cqk * Cout * 2);
  __bf16* wvb  = (__bf16*)alloc((size_t)Cout * Cout * 2);
  __bf16* wfb  = (__bf16*)alloc((size_t)Cout * Cout * 9 * 2);
  float*  sc1  = (float*)alloc(Cout * 4);
  float*  bi1  = (float*)alloc(Cout * 4);
  float*  sc2  = (float*)alloc(Cout * 4);
  float*  bi2  = (float*)alloc(Cout * 4);
  float*  x1f  = (float*)alloc((size_t)B * Cout * NSP * 4);
  __bf16* x1b  = (__bf16*)alloc((size_t)B * Cout * NSP * 2);
  __bf16* qb   = (__bf16*)alloc((size_t)B * Cqk * NSP * 2);
  __bf16* kb   = (__bf16*)alloc((size_t)B * Cqk * NSP * 2);
  __bf16* vb   = (__bf16*)alloc((size_t)B * Cout * NSP * 2);
  __bf16* y2b  = (__bf16*)alloc((size_t)B * Cout * NSP * 2);

  auto conv_blocks = [](long long n) { return (unsigned)((n + 255) / 256); };

  // ---- converts + BN folds ----
  k_f32_to_bf16<<<conv_blocks((long long)B * Cin * NSP), 256, 0, stream>>>(
      x, xb, (long long)B * Cin * NSP);
  k_f32_to_bf16<<<conv_blocks((long long)Cout * Cin * 9), 256, 0, stream>>>(
      w_pre, wpre, (long long)Cout * Cin * 9);
  k_f32_to_bf16<<<conv_blocks(Cqk * Cout), 256, 0, stream>>>(w_q, wqb, Cqk * Cout);
  k_f32_to_bf16<<<conv_blocks(Cqk * Cout), 256, 0, stream>>>(w_k, wkb, Cqk * Cout);
  k_f32_to_bf16<<<conv_blocks(Cout * Cout), 256, 0, stream>>>(w_v, wvb, Cout * Cout);
  k_f32_to_bf16<<<conv_blocks((long long)Cout * Cout * 9), 256, 0, stream>>>(
      w_f, wfb, (long long)Cout * Cout * 9);
  k_bn_fold<<<1, 256, 0, stream>>>(b_pre, bn1_g, bn1_b, bn1_m, bn1_v, sc1, bi1, Cout);
  k_bn_fold<<<1, 256, 0, stream>>>(b_f, bn2_g, bn2_b, bn2_m, bn2_v, sc2, bi2, Cout);

  // ---- pre_conv + BN + ReLU (fp32 + bf16 outputs) ----
  k_conv3x3_wmma<<<dim3(Cout / 128, NSP / 64, B), 256, 0, stream>>>(
      wpre, xb, sc1, bi1, x1f, x1b, Cin, Cout);

  // ---- q / k / v (1x1) ----
  k_gemm1x1_wmma<<<dim3(Cqk / 64, NSP / 64, B), 256, 0, stream>>>(
      wqb, x1b, b_q, qb, Cqk, Cout);
  k_gemm1x1_wmma<<<dim3(Cqk / 64, NSP / 64, B), 256, 0, stream>>>(
      wkb, x1b, b_k, kb, Cqk, Cout);
  k_gemm1x1_wmma<<<dim3(Cout / 64, NSP / 64, B), 256, 0, stream>>>(
      wvb, x1b, b_v, vb, Cout, Cout);

  // ---- fused softmax(q^T k) v + residual ----
  k_attn_flash<<<dim3(NSP / 32, B), 256, 0, stream>>>(qb, kb, vb, x1f, gamma, y2b);

  // ---- final_conv + BN + ReLU -> d_out (fp32) ----
  k_conv3x3_wmma<<<dim3(Cout / 128, NSP / 64, B), 256, 0, stream>>>(
      wfb, y2b, sc2, bi2, (float*)d_out, nullptr, Cout, Cout);
}
